// Block_51256139711130
// MI455X (gfx1250) — matrix operand order repaired and checked by reading, not compiled, not run on hardware
//
#include <hip/hip_runtime.h>
#include <math.h>

typedef __attribute__((ext_vector_type(16))) _Float16 v16h;
typedef __attribute__((ext_vector_type(8)))  _Float16 v8h;
typedef __attribute__((ext_vector_type(8)))  float    v8f;

#define DEV __device__ __forceinline__

#define CDIM   768
#define TDIM   16
#define HDIM   32
#define WDIM   64
#define NBLK   8
#define BSZ    96
#define NSEQ   2048     // H*W
#define NTOK   32768    // H*W*T
#define NFT    9        // T/2+1
#define NFW    33       // W/2+1
#define HIDDEN 3072

DEV v8f zero_v8f() {
    v8f z;
#pragma unroll
    for (int i = 0; i < 8; ++i) z[i] = 0.0f;
    return z;
}

// A-fragment (16x32 f16) from row-major [M][K] storage, stride lda (halfs).
// wave32 layout: lanes 0-15: M=lane, K = {k0..k0+7, k0+16..k0+23}
//                lanes 16-31: M=lane-16, K = {k0+8..k0+15, k0+24..k0+31}
DEV v16h frag_a(const _Float16* A, int lda, int k0) {
    int lane = threadIdx.x & 31;
    int r  = lane & 15;
    int kb = k0 + ((lane >> 4) << 3);
    union { v16h v; v8h h[2]; } u;
    u.h[0] = *(const v8h*)(A + r * lda + kb);
    u.h[1] = *(const v8h*)(A + r * lda + kb + 16);
    return u.v;
}

// B-fragment (32x16 f16) from B^T storage [N][K], stride ldb (halfs).
// lanes 0-15: N=lane, K = {k0..k0+7, k0+16..k0+23} ; lanes 16-31: N=lane-16, K = {k0+8..k0+15, k0+24..k0+31}
DEV v16h frag_b(const _Float16* Bt, int ldb, int k0) {
    int lane = threadIdx.x & 31;
    int n  = lane & 15;
    int kb = k0 + ((lane >> 4) << 3);
    union { v16h v; v8h h[2]; } u;
    u.h[0] = *(const v8h*)(Bt + n * ldb + kb);
    u.h[1] = *(const v8h*)(Bt + n * ldb + kb + 16);
    return u.v;
}

DEV v8f wmma16(v16h a, v16h b, v8f c) {
    return __builtin_amdgcn_wmma_f32_16x16x32_f16(false, a, false, b, (short)0, c, false, false);
}

// D/C layout: lanes 0-15: N=lane, VGPR j -> M=j ; lanes 16-31: N=lane-16, M=j+8
DEV void store_d_f16(v8f acc, _Float16* D, int ldd) {
    int lane = threadIdx.x & 31;
    int n  = lane & 15;
    int rb = (lane >> 4) << 3;
#pragma unroll
    for (int j = 0; j < 8; ++j) D[(rb + j) * ldd + n] = (_Float16)acc[j];
}

DEV float wave_sum(float v) {
#pragma unroll
    for (int o = 16; o > 0; o >>= 1) v += __shfl_xor(v, o, 32);
    return v;
}

DEV float sshrink(float v) {
    return v > 0.01f ? v - 0.01f : (v < -0.01f ? v + 0.01f : 0.0f);
}

// ---------------------------------------------------------------------------
// Weight convert + transpose: dst[b][c][r] = (f16) src[b][r][c]
__global__ void __launch_bounds__(256) convt_kernel(const float* __restrict__ src,
                                                    _Float16* __restrict__ dst,
                                                    int R, int C) {
    int b  = blockIdx.y;
    int id = blockIdx.x * 256 + threadIdx.x;
    if (id >= R * C) return;
    int r = id / C, c = id % C;
    dst[(size_t)b * R * C + (size_t)c * R + r] = (_Float16)src[(size_t)b * R * C + (size_t)r * C + c];
}

// ---------------------------------------------------------------------------
// Row LayerNorm: out[row] = (x-mu)*rsqrt(var+eps)*g + b ; one wave per row
__global__ void __launch_bounds__(256) ln_kernel(const float* __restrict__ x,
                                                 const float* __restrict__ g,
                                                 const float* __restrict__ bb,
                                                 float* __restrict__ out) {
    int wave = threadIdx.x >> 5, lane = threadIdx.x & 31;
    int row = blockIdx.x * 8 + wave;
    const float* xr = x + (size_t)row * CDIM;
    float s = 0.f, ss = 0.f;
    for (int c = lane; c < CDIM; c += 32) { float v = xr[c]; s += v; ss += v * v; }
    s = wave_sum(s); ss = wave_sum(ss);
    float mu  = s * (1.0f / CDIM);
    float inv = rsqrtf(ss * (1.0f / CDIM) - mu * mu + 1e-5f);
    float* orow = out + (size_t)row * CDIM;
    for (int c = lane; c < CDIM; c += 32) orow[c] = (xr[c] - mu) * inv * g[c] + bb[c];
}

// ---------------------------------------------------------------------------
// Forward time rFFT (ortho): G[f][seq][c] = (1/4) * sum_t h[seq][t][c] e^{-i2pi f t/16}
__global__ void __launch_bounds__(256) dft_t_fwd(const float* __restrict__ h,
                                                 float* __restrict__ Gr,
                                                 float* __restrict__ Gi) {
    __shared__ float tc[NFT * TDIM], tsn[NFT * TDIM];
    for (int i = threadIdx.x; i < NFT * TDIM; i += 256) {
        int f = i >> 4, t = i & 15;
        float th = 0.39269908169872414f * (float)(f * t);
        sincosf(th, &tsn[i], &tc[i]);
    }
    __syncthreads();
    size_t col = (size_t)blockIdx.x * 256 + threadIdx.x;  // exact: 2048*768 threads
    int seq = (int)(col / CDIM), c = (int)(col % CDIM);
    float xr[TDIM];
#pragma unroll
    for (int t = 0; t < TDIM; ++t) xr[t] = h[((size_t)seq * TDIM + t) * CDIM + c];
    for (int f = 0; f < NFT; ++f) {
        float ar = 0.f, ai = 0.f;
#pragma unroll
        for (int t = 0; t < TDIM; ++t) {
            ar += xr[t] * tc[f * 16 + t];
            ai -= xr[t] * tsn[f * 16 + t];
        }
        size_t g = ((size_t)f * NSEQ + seq) * CDIM + c;
        Gr[g] = ar * 0.25f;
        Gi[g] = ai * 0.25f;
    }
}

// Inverse time rFFT (ortho) + residual: h += irfft(Y)
__global__ void __launch_bounds__(256) dft_t_inv(const float* __restrict__ Gr,
                                                 const float* __restrict__ Gi,
                                                 float* __restrict__ h) {
    __shared__ float tc[NFT * TDIM], tsn[NFT * TDIM];
    for (int i = threadIdx.x; i < NFT * TDIM; i += 256) {
        int f = i >> 4, t = i & 15;
        float th = 0.39269908169872414f * (float)(f * t);
        sincosf(th, &tsn[i], &tc[i]);
    }
    __syncthreads();
    size_t col = (size_t)blockIdx.x * 256 + threadIdx.x;
    int seq = (int)(col / CDIM), c = (int)(col % CDIM);
    float yr[NFT], yi[NFT];
#pragma unroll
    for (int f = 0; f < NFT; ++f) {
        size_t g = ((size_t)f * NSEQ + seq) * CDIM + c;
        yr[f] = Gr[g]; yi[f] = Gi[g];
    }
    for (int t = 0; t < TDIM; ++t) {
        float v = yr[0] + ((t & 1) ? -yr[8] : yr[8]);
#pragma unroll
        for (int f = 1; f < 8; ++f)
            v += 2.0f * (yr[f] * tc[f * 16 + t] - yi[f] * tsn[f * 16 + t]);
        size_t d = ((size_t)seq * TDIM + t) * CDIM + c;
        h[d] = v * 0.25f + h[d];
    }
}

// ---------------------------------------------------------------------------
// Forward W rFFT: G[t][h][fw][c] = s2 * sum_w h[h][w][t][c] e^{-i2pi fw w/64}
__global__ void __launch_bounds__(256) dft_w_fwd(const float* __restrict__ h,
                                                 float* __restrict__ Gr,
                                                 float* __restrict__ Gi) {
    __shared__ float wc[NFW * WDIM], wsn[NFW * WDIM];
    for (int i = threadIdx.x; i < NFW * WDIM; i += 256) {
        int f = i / WDIM, w = i % WDIM;
        float th = 0.09817477042468103f * (float)(f * w);
        sincosf(th, &wsn[i], &wc[i]);
    }
    __syncthreads();
    int id = blockIdx.x * 256 + threadIdx.x;  // exact: 16*32*768
    int t = id / (HDIM * CDIM);
    int rem = id % (HDIM * CDIM);
    int hh = rem / CDIM, c = rem % CDIM;
    float xr[WDIM];
#pragma unroll
    for (int w = 0; w < WDIM; ++w)
        xr[w] = h[(((size_t)hh * WDIM + w) * TDIM + t) * CDIM + c];
    const float sc = 0.022097086912079608f;  // 1/sqrt(2048)
    for (int f = 0; f < NFW; ++f) {
        float ar = 0.f, ai = 0.f;
#pragma unroll
        for (int w = 0; w < WDIM; ++w) {
            ar += xr[w] * wc[f * WDIM + w];
            ai -= xr[w] * wsn[f * WDIM + w];
        }
        size_t g = (((size_t)t * HDIM + hh) * NFW + f) * CDIM + c;
        Gr[g] = ar * sc;
        Gi[g] = ai * sc;
    }
}

// In-place complex DFT over the H axis; sign=-1 forward, +1 inverse (no scale).
__global__ void __launch_bounds__(256) dft_h_kernel(float* __restrict__ Gr,
                                                    float* __restrict__ Gi,
                                                    float sign) {
    __shared__ float hc_[HDIM * HDIM], hs_[HDIM * HDIM];
    for (int i = threadIdx.x; i < HDIM * HDIM; i += 256) {
        int f = i / HDIM, hh = i % HDIM;
        float th = 0.19634954084936207f * (float)(f * hh);
        sincosf(th, &hs_[i], &hc_[i]);
    }
    __syncthreads();
    int id = blockIdx.x * 256 + threadIdx.x;  // exact: 16*33*768
    int t = id / (NFW * CDIM);
    int rem = id % (NFW * CDIM);
    int fw = rem / CDIM, c = rem % CDIM;
    float fr[HDIM], fi[HDIM];
#pragma unroll
    for (int hh = 0; hh < HDIM; ++hh) {
        size_t g = (((size_t)t * HDIM + hh) * NFW + fw) * CDIM + c;
        fr[hh] = Gr[g]; fi[hh] = Gi[g];
    }
    for (int fh = 0; fh < HDIM; ++fh) {
        float ar = 0.f, ai = 0.f;
#pragma unroll
        for (int hh = 0; hh < HDIM; ++hh) {
            float cc = hc_[fh * HDIM + hh], ss = hs_[fh * HDIM + hh];
            ar += fr[hh] * cc - sign * fi[hh] * ss;
            ai += fi[hh] * cc + sign * fr[hh] * ss;
        }
        size_t g = (((size_t)t * HDIM + fh) * NFW + fw) * CDIM + c;
        Gr[g] = ar; Gi[g] = ai;
    }
}

// Inverse W rFFT + double residual: h = irfft(Y) + h + x
__global__ void __launch_bounds__(256) idft_w(const float* __restrict__ Gr,
                                              const float* __restrict__ Gi,
                                              const float* __restrict__ x,
                                              float* __restrict__ h) {
    __shared__ float wc[NFW * WDIM], wsn[NFW * WDIM];
    for (int i = threadIdx.x; i < NFW * WDIM; i += 256) {
        int f = i / WDIM, w = i % WDIM;
        float th = 0.09817477042468103f * (float)(f * w);
        sincosf(th, &wsn[i], &wc[i]);
    }
    __syncthreads();
    int id = blockIdx.x * 256 + threadIdx.x;  // exact: 16*32*768
    int t = id / (HDIM * CDIM);
    int rem = id % (HDIM * CDIM);
    int hh = rem / CDIM, c = rem % CDIM;
    float yr[NFW], yi[NFW];
#pragma unroll
    for (int f = 0; f < NFW; ++f) {
        size_t g = (((size_t)t * HDIM + hh) * NFW + f) * CDIM + c;
        yr[f] = Gr[g]; yi[f] = Gi[g];
    }
    const float sc = 0.022097086912079608f;
    for (int w = 0; w < WDIM; ++w) {
        float v = yr[0] + ((w & 1) ? -yr[32] : yr[32]);
#pragma unroll
        for (int f = 1; f < 32; ++f)
            v += 2.0f * (yr[f] * wc[f * WDIM + w] - yi[f] * wsn[f * WDIM + w]);
        size_t d = (((size_t)hh * WDIM + w) * TDIM + t) * CDIM + c;
        h[d] = v * sc + h[d] + x[d];
    }
}

// ---------------------------------------------------------------------------
// Block-diagonal complex 2-layer spectral MLP, in place on (Gr, Gi).
// Grid: (rows/64, NBLK). 128 threads = 4 waves; wave w owns M-tile w.
__global__ void __launch_bounds__(128) spectral_kernel(float* __restrict__ Gr,
                                                       float* __restrict__ Gi,
                                                       const _Float16* __restrict__ w1T,
                                                       const float* __restrict__ b1,
                                                       const _Float16* __restrict__ w2T,
                                                       const float* __restrict__ b2) {
    const int blk  = blockIdx.y;
    const int row0 = blockIdx.x * 64;
    const int tid  = threadIdx.x;
    __shared__ __align__(16) _Float16 Ar[64 * BSZ], Ai[64 * BSZ];
    __shared__ __align__(16) _Float16 O1r[64 * BSZ], O1i[64 * BSZ];

    for (int idx = tid; idx < 64 * BSZ; idx += 128) {
        int r = idx / BSZ, c = idx % BSZ;
        size_t g = (size_t)(row0 + r) * CDIM + blk * BSZ + c;
        Ar[idx] = (_Float16)Gr[g];
        Ai[idx] = (_Float16)Gi[g];
    }
    __syncthreads();

    const int wave = tid >> 5;
    const int lane = tid & 31;
    const _Float16* W1r = w1T + (size_t)blk * (BSZ * BSZ);
    const _Float16* W1i = w1T + (size_t)(NBLK + blk) * (BSZ * BSZ);
    const _Float16* W2r = w2T + (size_t)blk * (BSZ * BSZ);
    const _Float16* W2i = w2T + (size_t)(NBLK + blk) * (BSZ * BSZ);

    // Layer 1: o1 = relu(A * W1 + b1) (complex)
    for (int nt = 0; nt < 6; ++nt) {
        v8f prr = zero_v8f(), pii = zero_v8f(), pir = zero_v8f(), pri = zero_v8f();
#pragma unroll
        for (int k = 0; k < 3; ++k) {
            v16h ar = frag_a(Ar + wave * 16 * BSZ, BSZ, k * 32);
            v16h ai = frag_a(Ai + wave * 16 * BSZ, BSZ, k * 32);
            v16h br = frag_b(W1r + nt * 16 * BSZ, BSZ, k * 32);
            v16h bi = frag_b(W1i + nt * 16 * BSZ, BSZ, k * 32);
            prr = wmma16(ar, br, prr);
            pii = wmma16(ai, bi, pii);
            pir = wmma16(ai, br, pir);
            pri = wmma16(ar, bi, pri);
        }
        int n = nt * 16 + (lane & 15);
        float br1 = b1[blk * BSZ + n];
        float bi1 = b1[NBLK * BSZ + blk * BSZ + n];
        v8f o1r, o1i;
#pragma unroll
        for (int j = 0; j < 8; ++j) {
            o1r[j] = fmaxf(prr[j] - pii[j] + br1, 0.0f);
            o1i[j] = fmaxf(pir[j] + pri[j] + bi1, 0.0f);
        }
        store_d_f16(o1r, O1r + wave * 16 * BSZ + nt * 16, BSZ);
        store_d_f16(o1i, O1i + wave * 16 * BSZ + nt * 16, BSZ);
    }
    __syncthreads();

    // Layer 2: y = softshrink(o1 * W2 + b2) (complex), written back in place
    for (int nt = 0; nt < 6; ++nt) {
        v8f prr = zero_v8f(), pii = zero_v8f(), pir = zero_v8f(), pri = zero_v8f();
#pragma unroll
        for (int k = 0; k < 3; ++k) {
            v16h ar = frag_a(O1r + wave * 16 * BSZ, BSZ, k * 32);
            v16h ai = frag_a(O1i + wave * 16 * BSZ, BSZ, k * 32);
            v16h br = frag_b(W2r + nt * 16 * BSZ, BSZ, k * 32);
            v16h bi = frag_b(W2i + nt * 16 * BSZ, BSZ, k * 32);
            prr = wmma16(ar, br, prr);
            pii = wmma16(ai, bi, pii);
            pir = wmma16(ai, br, pir);
            pri = wmma16(ar, bi, pri);
        }
        int n  = nt * 16 + (lane & 15);
        int rb = (lane >> 4) << 3;
        float br2 = b2[blk * BSZ + n];
        float bi2 = b2[NBLK * BSZ + blk * BSZ + n];
#pragma unroll
        for (int j = 0; j < 8; ++j) {
            int row = row0 + wave * 16 + rb + j;
            size_t g = (size_t)row * CDIM + blk * BSZ + n;
            Gr[g] = sshrink(prr[j] - pii[j] + br2);
            Gi[g] = sshrink(pir[j] + pri[j] + bi2);
        }
    }
}

// ---------------------------------------------------------------------------
// Fused LN2 + fc1 + exact GELU + fc2 + bias + residual. 32-token tiles.
// 256 threads = 8 waves. fc1T: [3072][768] f16, fc2T: [768][3072] f16.
__global__ void __launch_bounds__(256) mlp_kernel(const float* __restrict__ h3,
                                                  const float* __restrict__ g2,
                                                  const float* __restrict__ bb2,
                                                  const _Float16* __restrict__ fc1T,
                                                  const float* __restrict__ fc1b,
                                                  const _Float16* __restrict__ fc2T,
                                                  const float* __restrict__ fc2b,
                                                  float* __restrict__ out) {
    const int tile0 = blockIdx.x * 32;
    const int tid = threadIdx.x, wave = tid >> 5, lane = tid & 31;
    __shared__ __align__(16) _Float16 Ax[32 * CDIM];   // 48KB normalized acts
    __shared__ __align__(16) _Float16 Hc[32 * 128];    // 8KB hidden chunk

    // LN2: 4 rows per wave
    for (int rr = 0; rr < 4; ++rr) {
        int r = wave + rr * 8;
        const float* xr = h3 + (size_t)(tile0 + r) * CDIM;
        float s = 0.f, ss = 0.f;
        for (int c = lane; c < CDIM; c += 32) { float v = xr[c]; s += v; ss += v * v; }
        s = wave_sum(s); ss = wave_sum(ss);
        float mu  = s * (1.0f / CDIM);
        float inv = rsqrtf(ss * (1.0f / CDIM) - mu * mu + 1e-5f);
        for (int c = lane; c < CDIM; c += 32)
            Ax[r * CDIM + c] = (_Float16)((xr[c] - mu) * inv * g2[c] + bb2[c]);
    }
    __syncthreads();

    v8f acc2[12];
#pragma unroll
    for (int i = 0; i < 12; ++i) acc2[i] = zero_v8f();

    for (int hc = 0; hc < HIDDEN / 128; ++hc) {
        // GEMM1: this wave computes hid cols [hc*128 + wave*16, +16) for both M-tiles
        const _Float16* Bt1 = fc1T + (size_t)(hc * 128 + wave * 16) * CDIM;
#pragma unroll
        for (int m = 0; m < 2; ++m) {
            v8f a = zero_v8f();
            for (int k = 0; k < CDIM / 32; ++k) {
                v16h af = frag_a(Ax + m * 16 * CDIM, CDIM, k * 32);
                v16h bf = frag_b(Bt1, CDIM, k * 32);
                a = wmma16(af, bf, a);
            }
            int n = lane & 15;
            float bias = fc1b[hc * 128 + wave * 16 + n];
#pragma unroll
            for (int j = 0; j < 8; ++j) {
                float v = a[j] + bias;
                a[j] = 0.5f * v * (1.0f + erff(v * 0.7071067811865475f));
            }
            store_d_f16(a, Hc + m * 16 * 128 + wave * 16, 128);
        }
        __syncthreads();
        // GEMM2 partial: wave owns 6 output N-tiles
#pragma unroll
        for (int ntl = 0; ntl < 6; ++ntl) {
            int n0 = (wave * 6 + ntl) * 16;
            const _Float16* Bt2 = fc2T + (size_t)n0 * HIDDEN + hc * 128;
#pragma unroll
            for (int m = 0; m < 2; ++m) {
                v8f acc = acc2[ntl * 2 + m];
#pragma unroll
                for (int k = 0; k < 4; ++k) {
                    v16h af = frag_a(Hc + m * 16 * 128, 128, k * 32);
                    v16h bf = frag_b(Bt2, HIDDEN, k * 32);
                    acc = wmma16(af, bf, acc);
                }
                acc2[ntl * 2 + m] = acc;
            }
        }
        __syncthreads();
    }

    // Epilogue: += fc2 bias + residual
    int rb = (lane >> 4) << 3;
#pragma unroll
    for (int ntl = 0; ntl < 6; ++ntl) {
        int n = (wave * 6 + ntl) * 16 + (lane & 15);
        float bias = fc2b[n];
#pragma unroll
        for (int m = 0; m < 2; ++m) {
#pragma unroll
            for (int j = 0; j < 8; ++j) {
                int row = tile0 + m * 16 + rb + j;
                size_t d = (size_t)row * CDIM + n;
                out[d] = acc2[ntl * 2 + m][j] + bias + h3[d];
            }
        }
    }
}

// ---------------------------------------------------------------------------
extern "C" void kernel_launch(void* const* d_in, const int* in_sizes, int n_in,
                              void* d_out, int out_size, void* d_ws, size_t ws_size,
                              hipStream_t stream) {
    (void)in_sizes; (void)n_in; (void)out_size; (void)ws_size;
    const float* x       = (const float*)d_in[0];
    const float* n1g     = (const float*)d_in[1];
    const float* n1b     = (const float*)d_in[2];
    const float* t_w1    = (const float*)d_in[3];
    const float* t_b1    = (const float*)d_in[4];
    const float* t_w2    = (const float*)d_in[5];
    const float* t_b2    = (const float*)d_in[6];
    const float* s_w1    = (const float*)d_in[7];
    const float* s_b1    = (const float*)d_in[8];
    const float* s_w2    = (const float*)d_in[9];
    const float* s_b2    = (const float*)d_in[10];
    const float* n2g     = (const float*)d_in[11];
    const float* n2b     = (const float*)d_in[12];
    const float* fc1_w   = (const float*)d_in[13];
    const float* fc1_b   = (const float*)d_in[14];
    const float* fc2_w   = (const float*)d_in[15];
    const float* fc2_b   = (const float*)d_in[16];

    char* w = (char*)d_ws;
    float* hbuf = (float*)w;               w += (size_t)NTOK * CDIM * 4;          // 96 MB
    float* Gr   = (float*)w;               w += (size_t)NFT * NSEQ * CDIM * 4;    // 56.6 MB (reused for 2D: 51.9 MB)
    float* Gi   = (float*)w;               w += (size_t)NFT * NSEQ * CDIM * 4;
    _Float16* wt1T = (_Float16*)w;         w += (size_t)2 * NBLK * BSZ * BSZ * 2;
    _Float16* wt2T = (_Float16*)w;         w += (size_t)2 * NBLK * BSZ * BSZ * 2;
    _Float16* ws1T = (_Float16*)w;         w += (size_t)2 * NBLK * BSZ * BSZ * 2;
    _Float16* ws2T = (_Float16*)w;         w += (size_t)2 * NBLK * BSZ * BSZ * 2;
    _Float16* fc1T = (_Float16*)w;         w += (size_t)CDIM * HIDDEN * 2;
    _Float16* fc2T = (_Float16*)w;         w += (size_t)CDIM * HIDDEN * 2;

    // Weight conversion / transposition to f16 B^T layouts
    {
        dim3 g((BSZ * BSZ + 255) / 256, 2 * NBLK);
        convt_kernel<<<g, 256, 0, stream>>>(t_w1, wt1T, BSZ, BSZ);
        convt_kernel<<<g, 256, 0, stream>>>(t_w2, wt2T, BSZ, BSZ);
        convt_kernel<<<g, 256, 0, stream>>>(s_w1, ws1T, BSZ, BSZ);
        convt_kernel<<<g, 256, 0, stream>>>(s_w2, ws2T, BSZ, BSZ);
        dim3 g1((CDIM * HIDDEN + 255) / 256, 1);
        convt_kernel<<<g1, 256, 0, stream>>>(fc1_w, fc1T, CDIM, HIDDEN);
        convt_kernel<<<g1, 256, 0, stream>>>(fc2_w, fc2T, HIDDEN, CDIM);
    }

    // LN1
    ln_kernel<<<NTOK / 8, 256, 0, stream>>>(x, n1g, n1b, hbuf);

    // AFNO1D over T
    dft_t_fwd<<<(NSEQ * CDIM) / 256, 256, 0, stream>>>(hbuf, Gr, Gi);
    spectral_kernel<<<dim3((NFT * NSEQ) / 64, NBLK), 128, 0, stream>>>(Gr, Gi, wt1T, t_b1, wt2T, t_b2);
    dft_t_inv<<<(NSEQ * CDIM) / 256, 256, 0, stream>>>(Gr, Gi, hbuf);

    // AFNO2D over (H, W)
    dft_w_fwd<<<(TDIM * HDIM * CDIM) / 256, 256, 0, stream>>>(hbuf, Gr, Gi);
    dft_h_kernel<<<(TDIM * NFW * CDIM) / 256, 256, 0, stream>>>(Gr, Gi, -1.0f);
    spectral_kernel<<<dim3((TDIM * HDIM * NFW) / 64, NBLK), 128, 0, stream>>>(Gr, Gi, ws1T, s_b1, ws2T, s_b2);
    dft_h_kernel<<<(TDIM * NFW * CDIM) / 256, 256, 0, stream>>>(Gr, Gi, +1.0f);
    idft_w<<<(TDIM * HDIM * CDIM) / 256, 256, 0, stream>>>(Gr, Gi, x, hbuf);

    // Fused LN2 + MLP + residual
    mlp_kernel<<<NTOK / 32, 256, 0, stream>>>(hbuf, n2g, n2b, fc1T, fc1_b, fc2T, fc2_b, (float*)d_out);
}
